// Encoder_77627238908751
// MI455X (gfx1250) — matrix operand order repaired and checked by reading, compiled, not run on hardware
//
#include <hip/hip_runtime.h>
#include <hip/hip_bf16.h>

// ---------------- types & sizes ----------------
typedef _Float16 h16_t;
typedef __attribute__((ext_vector_type(16))) _Float16 v16h;
typedef __attribute__((ext_vector_type(8)))  _Float16 v8h;
typedef __attribute__((ext_vector_type(8)))  float    v8f;

#define HD   512
#define HD2  1024
#define HD3  1536
#define BB   32
#define SS   256
#define TT   16
#define SB   (SS*BB)   // 8192
#define TB   (TT*BB)   // 512
#define LDW  520       // padded LDS row stride (halves), 16B aligned
#define NSLICE 8       // workgroups per GRU direction (64 H-cols each)

__device__ __forceinline__ v8f wmma_f16(v16h a, v16h b, v8f c) {
  // D(f32,16x16) = A(f16,16x32) x B(f16,32x16) + C
  return __builtin_amdgcn_wmma_f32_16x16x32_f16(
      /*neg_a=*/false, a, /*neg_b=*/false, b,
      /*c_mod=*/(short)0, c, /*reuse_a=*/false, /*reuse_b=*/false);
}

// ---------------- conversion / gather ----------------
__global__ void k_cvt_rows_f16(h16_t* __restrict__ dst, const float* __restrict__ src,
                               int rows, int cols, int src_ld, int col0) {
  int i = blockIdx.x * blockDim.x + threadIdx.x;
  if (i >= rows * cols) return;
  int r = i / cols, c = i - r * cols;
  dst[i] = (h16_t)src[(size_t)r * src_ld + col0 + c];
}

__global__ void k_gather_f16(h16_t* __restrict__ dst, const float* __restrict__ tab,
                             const int* __restrict__ idx, int nrows) {
  int i = blockIdx.x * blockDim.x + threadIdx.x;
  if (i >= nrows * HD) return;
  int r = i >> 9, c = i & (HD - 1);
  dst[i] = (h16_t)tab[(size_t)idx[r] * HD + c];
}

__global__ void k_init_sync(h16_t* __restrict__ hbuf, unsigned* __restrict__ sync) {
  int i = blockIdx.x * blockDim.x + threadIdx.x;
  if (i < 2 * 2 * BB * HD) hbuf[i] = (h16_t)0.f;   // h(0) = 0, both parities
  if (i < 128) sync[i] = 0u;                       // grid-barrier counters
}

// ---------------- generic WMMA GEMM: C[M,N] = A[M,K] * W[N,K]^T + bias[N] ----
// One wave -> 16x64 output tile (A-frag reused across 4 B-frags).
__global__ __launch_bounds__(256) void k_gemm_f16w(
    const h16_t* __restrict__ A, const h16_t* __restrict__ W,
    const float* __restrict__ bias, float* __restrict__ C,
    int M, int N, int K) {
  int wave = (blockIdx.x * blockDim.x + threadIdx.x) >> 5;
  int lane = threadIdx.x & 31;
  int r = lane & 15, hi = lane >> 4;
  int ngrp = N >> 6;
  int mt = wave / ngrp;
  int ng = wave - mt * ngrp;
  int m0 = mt << 4, n0 = ng << 6;

  v8f acc0 = {}, acc1 = {}, acc2 = {}, acc3 = {};
  const h16_t* arow = A + (size_t)(m0 + r) * K + hi * 16;
  const h16_t* wrow = W + (size_t)(n0 + r) * K + hi * 16;
  const size_t wstep = (size_t)16 * K;

  for (int k0 = 0; k0 < K; k0 += 32) {
    union { v16h v; v8h h[2]; } af;
    af.h[0] = *(const v8h*)(arow + k0);
    af.h[1] = *(const v8h*)(arow + k0 + 8);
    v16h b0 = *(const v16h*)(wrow + k0);
    v16h b1 = *(const v16h*)(wrow + k0 + wstep);
    v16h b2 = *(const v16h*)(wrow + k0 + 2 * wstep);
    v16h b3 = *(const v16h*)(wrow + k0 + 3 * wstep);
    acc0 = wmma_f16(af.v, b0, acc0);
    acc1 = wmma_f16(af.v, b1, acc1);
    acc2 = wmma_f16(af.v, b2, acc2);
    acc3 = wmma_f16(af.v, b3, acc3);
  }
  float bR0 = bias[n0 + r], bR1 = bias[n0 + 16 + r];
  float bR2 = bias[n0 + 32 + r], bR3 = bias[n0 + 48 + r];
  int rowb = m0 + hi * 8;
  #pragma unroll
  for (int e = 0; e < 8; ++e) {
    float* cp = C + (size_t)(rowb + e) * N + r;
    cp[n0]      = acc0[e] + bR0;
    cp[n0 + 16] = acc1[e] + bR1;
    cp[n0 + 32] = acc2[e] + bR2;
    cp[n0 + 48] = acc3[e] + bR3;
  }
}

// ---------------- bidirectional GRU scan, 16 workgroups ----------------
// grid.x = 16: dir = wg>>3, slice = wg&7 owns H-cols [slice*64, slice*64+64).
// 256 threads = 8 waves; wave -> unit (mt = wave&1 batch tile, chunk = wave>>1).
// h(t-1) (32x512 f16) is LDS-resident; each WG streams only its 192 Whh rows
// (~196KB, cache-hot in its WGP) per step instead of the full 1.57MB.
// h slices are exchanged through a double-buffered global f16 buffer with one
// device-scope atomic grid barrier per step (per direction).
__global__ __launch_bounds__(256) void k_gru_scan(
    const h16_t* __restrict__ Whhf16, const h16_t* __restrict__ Whhb16,
    const float* __restrict__ bhh_f, const float* __restrict__ bhh_b,
    const float* __restrict__ gx_f, const float* __restrict__ gx_b,
    h16_t* __restrict__ hbuf, unsigned* __restrict__ sync,
    float* __restrict__ enc_cat, h16_t* __restrict__ enc_cat16) {
  __shared__ h16_t hsh[BB * LDW];
  __shared__ unsigned done_flag;

  const int wg = blockIdx.x;
  const int dir = wg >> 3;
  const int slice = wg & 7;
  const h16_t* Whh = dir ? Whhb16 : Whhf16;
  const float* bhh = dir ? bhh_b : bhh_f;
  const float* gx  = dir ? gx_b  : gx_f;
  h16_t* hb = hbuf + (size_t)dir * 2 * (BB * HD);
  unsigned* ctr = sync + dir * 64;   // separate cacheline per direction

  const int tid = threadIdx.x;
  const int wave = tid >> 5, lane = tid & 31;
  const int r = lane & 15, hi = lane >> 4;
  const int mt = wave & 1;                  // batch tile (rows mt*16..mt*16+15)
  const int c16 = slice * 4 + (wave >> 1);  // global 16-col chunk id (0..31)
  const int cc = (c16 << 4) + r;            // this lane's H column

  const h16_t* brp = Whh + (size_t)cc * HD + hi * 16;  // gate-r row for lane

  // fill LDS with h(0) = 0 from hbuf parity 0
  for (int i = tid; i < (BB * HD) / 8; i += 256) {
    int row = i >> 6, seg = i & 63;
    *(v8h*)&hsh[row * LDW + seg * 8] = *(const v8h*)&hb[(size_t)row * HD + seg * 8];
  }
  __syncthreads();

  for (int step = 0; step < SS; ++step) {
    const int s = dir ? (SS - 1 - step) : step;
    const int par = step & 1;       // LDS holds h from hbuf[par]

    // prefetch the gx gate rows this wave will need after the GEMM
    __builtin_prefetch(&gx[((size_t)s * BB + mt * 16 + hi * 8) * HD3 + cc], 0, 1);

    // --- h(t-1) @ Whh^T for this wave's (mt, chunk): 3 gate chains over K=512
    v8f ar = {}, az = {}, an = {};
    const h16_t* hrow = &hsh[(mt * 16 + r) * LDW + hi * 16];
    for (int k0 = 0; k0 < HD; k0 += 32) {
      union { v16h v; v8h h[2]; } af;
      af.h[0] = *(const v8h*)(hrow + k0);
      af.h[1] = *(const v8h*)(hrow + k0 + 8);
      ar = wmma_f16(af.v, *(const v16h*)(brp + k0), ar);
      az = wmma_f16(af.v, *(const v16h*)(brp + (size_t)HD  * HD + k0), az);
      an = wmma_f16(af.v, *(const v16h*)(brp + (size_t)HD2 * HD + k0), an);
    }

    // --- gate math + writes (hbuf[par^1] slice, enc_cat f32/f16)
    const float bR = bhh[cc], bZ = bhh[HD + cc], bN = bhh[HD2 + cc];
    h16_t* hout = hb + (size_t)(par ^ 1) * (BB * HD);
    #pragma unroll
    for (int e = 0; e < 8; ++e) {
      int row = mt * 16 + hi * 8 + e;
      size_t g = ((size_t)s * BB + row) * HD3;
      float rg = 1.f / (1.f + __expf(-(gx[g + cc]      + ar[e] + bR)));
      float zg = 1.f / (1.f + __expf(-(gx[g + HD + cc] + az[e] + bZ)));
      float ng = tanhf(gx[g + HD2 + cc] + rg * (an[e] + bN));
      float hp = (float)hsh[row * LDW + cc];
      float hv = (1.f - zg) * ng + zg * hp;
      hout[(size_t)row * HD + cc] = (h16_t)hv;
      size_t ob = ((size_t)s * BB + row) * HD2 + (size_t)dir * HD + cc;
      enc_cat[ob]   = hv;
      enc_cat16[ob] = (h16_t)hv;
    }

    // --- device-scope grid barrier (8 WGs of this direction)
    __threadfence();
    __syncthreads();
    if (tid == 0) {
      unsigned target = (unsigned)(NSLICE * (step + 1));
      __hip_atomic_fetch_add(ctr, 1u, __ATOMIC_ACQ_REL, __HIP_MEMORY_SCOPE_AGENT);
      while (__hip_atomic_load(ctr, __ATOMIC_ACQUIRE, __HIP_MEMORY_SCOPE_AGENT) < target)
        __builtin_amdgcn_s_sleep(2);
      done_flag = target;   // defeat any hoisting; makes exit data-dependent
    }
    __syncthreads();

    // --- refill LDS with full h(t) from hbuf[par^1]
    const h16_t* hin = hb + (size_t)(par ^ 1) * (BB * HD);
    for (int i = tid; i < (BB * HD) / 8; i += 256) {
      int row = i >> 6, seg = i & 63;
      *(v8h*)&hsh[row * LDW + seg * 8] = *(const v8h*)&hin[(size_t)row * HD + seg * 8];
    }
    __syncthreads();
  }
  if (tid == 0 && done_flag == 0xffffffffu) enc_cat[0] = 0.f;  // never taken
}

// ---------------- small f32 kernels ----------------
__global__ void k_theme_proj(const int* __restrict__ theme, const float* __restrict__ tab,
                             const float* __restrict__ Wt, const float* __restrict__ bt,
                             float* __restrict__ tp) {
  int i = blockIdx.x * blockDim.x + threadIdx.x;  // 32*512
  int b = i >> 9, h = i & (HD - 1);
  const float* e = tab + (size_t)theme[b] * HD;
  const float* w = Wt + (size_t)h * HD;
  float a = bt[h];
  for (int k = 0; k < HD; ++k) a += e[k] * w[k];
  tp[i] = a;
}

__global__ void k_tp1(const float* __restrict__ tp, const float* __restrict__ Wo,
                      float* __restrict__ tp1) {
  int i = blockIdx.x * blockDim.x + threadIdx.x;  // 32*512
  int b = i >> 9, h = i & (HD - 1);
  const float* x = tp + (size_t)b * HD;
  const float* w = Wo + (size_t)h * HD3;          // Wo[h, 0:H]
  float a = 0.f;
  for (int k = 0; k < HD; ++k) a += x[k] * w[k];
  tp1[i] = a;
}

// sumt[s*B+b] = sum_t sum_h tanh(b1[sb,h] + tp[b,h] + b3[t*B+b,h]) * wv[h] + T*bv
__global__ __launch_bounds__(256) void k_scores(
    const float* __restrict__ b1, const float* __restrict__ tp,
    const float* __restrict__ b3, const float* __restrict__ wv,
    const float* __restrict__ bv, float* __restrict__ sumt) {
  __shared__ float csh[HD];
  __shared__ float red[256];
  int sb = blockIdx.x;
  int b = sb & (BB - 1);
  int tid = threadIdx.x;
  int h0 = tid, h1 = tid + 256;
  csh[h0] = b1[(size_t)sb * HD + h0] + tp[(size_t)b * HD + h0];
  csh[h1] = b1[(size_t)sb * HD + h1] + tp[(size_t)b * HD + h1];
  __syncthreads();
  float w0 = wv[h0], w1 = wv[h1];
  float acc = 0.f;
  #pragma unroll 4
  for (int t = 0; t < TT; ++t) {
    const float* b3r = b3 + (size_t)(t * BB + b) * HD;
    acc += tanhf(csh[h0] + b3r[h0]) * w0;
    acc += tanhf(csh[h1] + b3r[h1]) * w1;
  }
  red[tid] = acc;
  __syncthreads();
  for (int st = 128; st > 0; st >>= 1) {
    if (tid < st) red[tid] += red[tid + st];
    __syncthreads();
  }
  if (tid == 0) sumt[sb] = red[0] + (float)TT * bv[0];
}

// enc_out[sb,h] += sumt[sb] * tp1[b,h]   (enc_out already holds enc@Wo2^T + bo)
__global__ void k_final(float* __restrict__ out, const float* __restrict__ sumt,
                        const float* __restrict__ tp1) {
  int i = blockIdx.x * blockDim.x + threadIdx.x;
  if (i >= SB * HD) return;
  int sb = i >> 9, h = i & (HD - 1), b = sb & (BB - 1);
  out[i] += sumt[sb] * tp1[(size_t)b * HD + h];
}

// enc_hidden_out[b,h] = dot(concat(ys_f[S-1,b], ys_b[0,b]), Who[h,:])
__global__ void k_hidden(const float* __restrict__ enc_cat, const float* __restrict__ Who,
                         float* __restrict__ out2) {
  int i = blockIdx.x * blockDim.x + threadIdx.x;  // 32*512
  int b = i >> 9, h = i & (HD - 1);
  const float* hf = enc_cat + ((size_t)(SS - 1) * BB + b) * HD2;      // fwd last
  const float* hb = enc_cat + (size_t)b * HD2 + HD;                   // bwd at s=0
  const float* w  = Who + (size_t)h * HD2;
  float a = 0.f;
  for (int k = 0; k < HD; ++k) a += hf[k] * w[k] + hb[k] * w[HD + k];
  out2[i] = a;
}

// ---------------- host ----------------
extern "C" void kernel_launch(void* const* d_in, const int* in_sizes, int n_in,
                              void* d_out, int out_size, void* d_ws, size_t ws_size,
                              hipStream_t stream) {
  (void)in_sizes; (void)n_in; (void)out_size; (void)ws_size;
  const int*   theme   = (const int*)d_in[0];
  const int*   keyword = (const int*)d_in[1];
  const int*   src     = (const int*)d_in[2];
  const float* theme_tab   = (const float*)d_in[3];
  const float* keyword_tab = (const float*)d_in[4];
  const float* src_tab     = (const float*)d_in[5];
  const float* Wih_f = (const float*)d_in[6];
  const float* Whh_f = (const float*)d_in[7];
  const float* bih_f = (const float*)d_in[8];
  const float* bhh_f = (const float*)d_in[9];
  const float* Wih_b = (const float*)d_in[10];
  const float* Whh_b = (const float*)d_in[11];
  const float* bih_b = (const float*)d_in[12];
  const float* bhh_b = (const float*)d_in[13];
  const float* Ww  = (const float*)d_in[14];
  const float* bw  = (const float*)d_in[15];
  const float* Wt  = (const float*)d_in[16];
  const float* bt  = (const float*)d_in[17];
  const float* Wk  = (const float*)d_in[18];
  const float* bk  = (const float*)d_in[19];
  const float* wv  = (const float*)d_in[20];
  const float* bv  = (const float*)d_in[21];
  const float* Wo  = (const float*)d_in[22];
  const float* bo  = (const float*)d_in[23];
  const float* Who = (const float*)d_in[24];

  char* base = (char*)d_ws;
  size_t off = 0;
  auto alloc = [&](size_t bytes) -> void* {
    off = (off + 255) & ~(size_t)255;
    void* p = base + off;
    off += bytes;
    return p;
  };
  h16_t* Wihf16 = (h16_t*)alloc((size_t)HD3 * HD * 2);
  h16_t* Whhf16 = (h16_t*)alloc((size_t)HD3 * HD * 2);
  h16_t* Wihb16 = (h16_t*)alloc((size_t)HD3 * HD * 2);
  h16_t* Whhb16 = (h16_t*)alloc((size_t)HD3 * HD * 2);
  h16_t* Ww16   = (h16_t*)alloc((size_t)HD  * HD2 * 2);
  h16_t* Wk16   = (h16_t*)alloc((size_t)HD  * HD * 2);
  h16_t* Wo216  = (h16_t*)alloc((size_t)HD  * HD2 * 2);   // Wo[:, H:3H]
  h16_t* srcemb16 = (h16_t*)alloc((size_t)SB * HD * 2);
  h16_t* kwemb16  = (h16_t*)alloc((size_t)TB * HD * 2);
  float* gxf    = (float*)alloc((size_t)SB * HD3 * 4);
  float* gxb    = (float*)alloc((size_t)SB * HD3 * 4);
  float* enc_cat   = (float*)alloc((size_t)SB * HD2 * 4);
  h16_t* enc_cat16 = (h16_t*)alloc((size_t)SB * HD2 * 2);
  float* b1buf  = (float*)alloc((size_t)SB * HD * 4);
  float* b3buf  = (float*)alloc((size_t)TB * HD * 4);
  float* tp     = (float*)alloc((size_t)BB * HD * 4);
  float* tp1    = (float*)alloc((size_t)BB * HD * 4);
  float* sumt   = (float*)alloc((size_t)SB * 4);
  h16_t* hbuf   = (h16_t*)alloc((size_t)2 * 2 * BB * HD * 2);  // dir x parity x 32x512
  unsigned* syncb = (unsigned*)alloc(128 * sizeof(unsigned));

  auto cvt = [&](h16_t* dst, const float* s, int rows, int cols, int ld, int c0) {
    int n = rows * cols;
    k_cvt_rows_f16<<<(n + 255) / 256, 256, 0, stream>>>(dst, s, rows, cols, ld, c0);
  };
  auto gemm = [&](const h16_t* A, const h16_t* W, const float* bias, float* C,
                  int M, int N, int K) {
    int waves = (M / 16) * (N / 64);
    k_gemm_f16w<<<waves / 8, 256, 0, stream>>>(A, W, bias, C, M, N, K);
  };

  // 1) weight + embedding conversion to f16; barrier state init
  cvt(Wihf16, Wih_f, HD3, HD, HD, 0);
  cvt(Whhf16, Whh_f, HD3, HD, HD, 0);
  cvt(Wihb16, Wih_b, HD3, HD, HD, 0);
  cvt(Whhb16, Whh_b, HD3, HD, HD, 0);
  cvt(Ww16, Ww, HD, HD2, HD2, 0);
  cvt(Wk16, Wk, HD, HD, HD, 0);
  cvt(Wo216, Wo, HD, HD2, HD3, HD);
  k_gather_f16<<<(SB * HD) / 256, 256, 0, stream>>>(srcemb16, src_tab, src, SB);
  k_gather_f16<<<(TB * HD) / 256, 256, 0, stream>>>(kwemb16, keyword_tab, keyword, TB);
  k_init_sync<<<(2 * 2 * BB * HD + 255) / 256, 256, 0, stream>>>(hbuf, syncb);

  // 2) GRU input projections (big parallel WMMA GEMMs, bias = bih)
  gemm(srcemb16, Wihf16, bih_f, gxf, SB, HD3, HD);
  gemm(srcemb16, Wihb16, bih_b, gxb, SB, HD3, HD);

  // 3) theme projection + its Wo1 image (tiny)
  k_theme_proj<<<(BB * HD) / 256, 256, 0, stream>>>(theme, theme_tab, Wt, bt, tp);
  k_tp1<<<(BB * HD) / 256, 256, 0, stream>>>(tp, Wo, tp1);

  // 4) bidirectional GRU scan: 16 cooperating WGs, atomic grid barrier per step
  k_gru_scan<<<2 * NSLICE, 256, 0, stream>>>(Whhf16, Whhb16, bhh_f, bhh_b,
                                             gxf, gxb, hbuf, syncb,
                                             enc_cat, enc_cat16);

  // 5) attention GEMMs
  gemm(enc_cat16, Ww16, bw, b1buf, SB, HD, HD2);      // b1 (rows s*B+b)
  gemm(kwemb16,  Wk16, bk, b3buf, TB, HD, HD);        // b3 (rows t*B+b)

  // 6) fused tanh-score reduction -> sumt (never materializes (B,S,T,H))
  k_scores<<<SB, 256, 0, stream>>>(b1buf, tp, b3buf, wv, bv, sumt);

  // 7) enc_out = enc@Wo2^T + bo  (+= sumt * tp1 afterwards)
  gemm(enc_cat16, Wo216, bo, (float*)d_out, SB, HD, HD2);
  k_final<<<(SB * HD) / 256, 256, 0, stream>>>((float*)d_out, sumt, tp1);

  // 8) enc_hidden_out
  k_hidden<<<(BB * HD) / 256, 256, 0, stream>>>(enc_cat, Who,
                                                (float*)d_out + (size_t)SB * HD);
}
